// GAT_23605140258859
// MI455X (gfx1250) — compile-verified, matrix operand order checked by reading, not run on hardware
//
#include <hip/hip_runtime.h>

// ---------------------------------------------------------------------------
// GAT forward on gfx1250 (MI455X), wave32 + WMMA + async global->LDS staging.
// B=4, N=2048, F_in=128, H=8, F_hid=64, C=32.
// Flash-attention-style fused masked softmax (never materializes the
// [B,H,N,N] score matrix), f16 operands / f32 WMMA accumulation.
// Attention B-tiles are double-buffered into LDS with
// GLOBAL_LOAD_ASYNC_TO_LDS_B128 + s_wait_asynccnt (ASYNCcnt domain).
// ---------------------------------------------------------------------------

#define B_   4
#define N_   2048
#define FIN  128
#define H_   8
#define FH   64
#define C_   32
#define HF   (H_ * FH)          // 512
#define ALPHA_    0.2f
#define OUTSLOPE_ 0.01f
#define NEGINF_   (-9e15f)

typedef __attribute__((ext_vector_type(16))) _Float16 v16h;
typedef __attribute__((ext_vector_type(8)))  _Float16 v8h;
typedef __attribute__((ext_vector_type(8)))  float    v8f;
typedef __attribute__((ext_vector_type(4)))  float    v4f;

union AFrag { v16h v; v8h p[2]; _Float16 h[16]; };

__device__ __forceinline__ v8f wmma_f16(v16h a, v16h b, v8f c) {
  // D = A(16x32 f16) * B(32x16 f16) + C(16x16 f32)
  return __builtin_amdgcn_wmma_f32_16x16x32_f16(false, a, false, b,
                                                (short)0, c, false, false);
}

// LDS byte offset (addrspace(3)) of a generic pointer into __shared__ memory,
// as required by the VDST operand of global_load_async_to_lds_*.
__device__ __forceinline__ unsigned lds_addr(const void* p) {
  return (unsigned)(unsigned long long)
      (__attribute__((address_space(3))) const char*)p;
}

// Async copy 16B/lane from global to LDS; tracked by ASYNCcnt.
__device__ __forceinline__ void async_b128(unsigned lds_off, const void* gaddr) {
  asm volatile("global_load_async_to_lds_b128 %0, %1, off"
               :: "v"(lds_off), "v"(gaddr) : "memory");
}
__device__ __forceinline__ void wait_async_le8() {
  asm volatile("s_wait_asynccnt 0x8" ::: "memory");
}
__device__ __forceinline__ void wait_async_0() {
  asm volatile("s_wait_asynccnt 0x0" ::: "memory");
}
__device__ __forceinline__ void wait_ds_0() {
  asm volatile("s_wait_dscnt 0x0" ::: "memory");
}

// A/B fragment layout (wave32, 16-bit, 16x32 / 32x16):
//   lane: half = lane>>4, row/col = lane&15, kbase = half*8
//   half index s in v16h -> K = kbase + (s<8 ? s : s+8)
// C/D f32 16x16: VGPR r, lane<16 -> (M=r, N=lane); lane>=16 -> (M=r+8, N=lane-16)

// ---------------------------------------------------------------------------
// Kernel 1: Wh = x @ W per (b,h), emitted transposed as WhT[bh][o][node] f16.
// LDS-staged f32->f16 tiles; padded pitch keeps ds_load_b128 16B-aligned.
// ---------------------------------------------------------------------------
#define XP 136  // LDS row pitch in halves: 272B, multiple of 16B, conflict-spreading

__global__ __launch_bounds__(256) void k_wh(const float* __restrict__ x,
                                            const float* __restrict__ W,
                                            _Float16* __restrict__ whT) {
  __shared__ __align__(16) _Float16 xs[64 * XP];
  __shared__ __align__(16) _Float16 wsT[64 * XP];
  const int n0 = blockIdx.x * 64;
  const int bh = blockIdx.y;
  const int b = bh >> 3, h = bh & 7;
  const int tid = threadIdx.x;

  for (int i = tid; i < 64 * FIN; i += 256) {          // x tile, row-major
    int r = i >> 7, k = i & 127;
    xs[r * XP + k] = (_Float16)x[((size_t)b * N_ + n0 + r) * FIN + k];
  }
  for (int i = tid; i < FIN * FH; i += 256) {          // W tile, transposed
    int k = i >> 6, o = i & 63;
    wsT[o * XP + k] = (_Float16)W[((size_t)h * FIN + k) * FH + o];
  }
  __syncthreads();

  const int wave = tid >> 5, lane = tid & 31;
  const int half = lane >> 4, mrow = lane & 15, kbase = half * 8;
  const int mt = wave & 3;                 // 4 row tiles of 16
  const int nt0 = (wave >> 2) * 2;         // 2 col tiles of 16 per wave
  v8f acc0 = {}, acc1 = {};
#pragma unroll
  for (int ks = 0; ks < 4; ++ks) {         // K = 128 = 4 x 32
    const int k0 = ks * 32 + kbase;
    AFrag a, b0, b1;
    a.p[0]  = *(const v8h*)&xs[(mt * 16 + mrow) * XP + k0];
    a.p[1]  = *(const v8h*)&xs[(mt * 16 + mrow) * XP + k0 + 16];
    b0.p[0] = *(const v8h*)&wsT[(nt0 * 16 + mrow) * XP + k0];
    b0.p[1] = *(const v8h*)&wsT[(nt0 * 16 + mrow) * XP + k0 + 16];
    b1.p[0] = *(const v8h*)&wsT[((nt0 + 1) * 16 + mrow) * XP + k0];
    b1.p[1] = *(const v8h*)&wsT[((nt0 + 1) * 16 + mrow) * XP + k0 + 16];
    acc0 = wmma_f16(a.v, b0.v, acc0);
    acc1 = wmma_f16(a.v, b1.v, acc1);
  }
#pragma unroll
  for (int r = 0; r < 8; ++r) {
    const int node = n0 + mt * 16 + r + 8 * half;
    whT[((size_t)bh * FH + nt0 * 16 + mrow) * N_ + node]       = (_Float16)acc0[r];
    whT[((size_t)bh * FH + (nt0 + 1) * 16 + mrow) * N_ + node] = (_Float16)acc1[r];
  }
}

// ---------------------------------------------------------------------------
// Kernel 2: f1/f2 = Wh . a1/a2  (coalesced over WhT columns)
// ---------------------------------------------------------------------------
__global__ __launch_bounds__(256) void k_f12(const _Float16* __restrict__ whT,
                                             const float* __restrict__ a1,
                                             const float* __restrict__ a2,
                                             float* __restrict__ f1,
                                             float* __restrict__ f2) {
  const int bh = blockIdx.y, h = bh & 7;
  const int n = blockIdx.x * 256 + threadIdx.x;
  float s1 = 0.f, s2 = 0.f;
#pragma unroll 8
  for (int o = 0; o < FH; ++o) {
    const float v = (float)whT[((size_t)bh * FH + o) * N_ + n];
    s1 += v * a1[h * FH + o];
    s2 += v * a2[h * FH + o];
  }
  f1[(size_t)bh * N_ + n] = s1;
  f2[(size_t)bh * N_ + n] = s2;
}

// ---------------------------------------------------------------------------
// Kernel 3: layer-1 fused masked softmax + P@Wh, ELU, head-concat.
// Block = (b, 16-row tile); wave w = head w so the adj tile is reused 8x
// in-cache. Wh B-tiles (64x32 f16) are double-buffered into wave-private LDS
// via GLOBAL_LOAD_ASYNC_TO_LDS_B128 (8 x 512B transfers per tile).
// ---------------------------------------------------------------------------
#define TP 48   // LDS tile pitch in halves: 96B = 6x16B aligned, spreads banks

__global__ __launch_bounds__(256) void k_attn1(const _Float16* __restrict__ whT,
                                               const float* __restrict__ adj,
                                               const float* __restrict__ f1,
                                               const float* __restrict__ f2,
                                               _Float16* __restrict__ hcat) {
  __shared__ __align__(16) _Float16 tiles[H_ * 2 * 64 * TP];   // 96 KB of 320 KB
  const int b = blockIdx.y;
  const int row0 = blockIdx.x * 16;
  const int h = threadIdx.x >> 5;          // wave == head
  const int lane = threadIdx.x & 31;
  const int half = lane >> 4, mrow = lane & 15, kbase = half * 8;
  const int bh = b * H_ + h;
  const int myrow = row0 + mrow;

  const float f1v = f1[(size_t)bh * N_ + myrow];
  const float* __restrict__ adjr = adj + ((size_t)b * N_ + myrow) * N_;
  const float* __restrict__ f2r  = f2 + (size_t)bh * N_;
  const _Float16* __restrict__ whTh = whT + (size_t)bh * FH * N_;
  _Float16* mytile = &tiles[h * 2 * 64 * TP];

  const int so = lane >> 2;                // staging: 4 lanes per row
  const int sseg = lane & 3;               // 16B segment within 64B row chunk

  auto stage = [&](int j0, int bufsel) {   // async copy WhT[:, j0:j0+32]
    _Float16* dst = mytile + bufsel * 64 * TP;
#pragma unroll
    for (int c = 0; c < 8; ++c) {
      const int o = c * 8 + so;
      async_b128(lds_addr(dst + o * TP + sseg * 8),
                 whTh + (size_t)o * N_ + j0 + sseg * 8);
    }
  };

  v8f acc[4] = {{}, {}, {}, {}};
  float runmax = -INFINITY, runsum = 0.f;

  stage(0, 0);                             // prologue: prefetch first tile
  for (int j0 = 0; j0 < N_; j0 += 32) {
    const int cur = (j0 >> 5) & 1;
    if (j0 + 32 < N_) {
      wait_ds_0();                         // WAR: previous reads of buf cur^1 done
      stage(j0 + 32, cur ^ 1);
      wait_async_le8();                    // tile `cur` complete (in-order)
    } else {
      wait_async_0();
    }

    const int cb = j0 + kbase;
    __builtin_prefetch((const void*)(adjr + j0 + 64), 0, 0);  // global_prefetch
    float av[16], fv[16];
    *(v4f*)&av[0]  = *(const v4f*)(adjr + cb);
    *(v4f*)&av[4]  = *(const v4f*)(adjr + cb + 4);
    *(v4f*)&av[8]  = *(const v4f*)(adjr + cb + 16);
    *(v4f*)&av[12] = *(const v4f*)(adjr + cb + 20);
    *(v4f*)&fv[0]  = *(const v4f*)(f2r + cb);
    *(v4f*)&fv[4]  = *(const v4f*)(f2r + cb + 4);
    *(v4f*)&fv[8]  = *(const v4f*)(f2r + cb + 16);
    *(v4f*)&fv[12] = *(const v4f*)(f2r + cb + 20);

    float ev[16], cmax = -INFINITY;
#pragma unroll
    for (int s = 0; s < 16; ++s) {
      float t = f1v + fv[s];
      t = t > 0.f ? t : ALPHA_ * t;
      ev[s] = av[s] > 0.f ? t : NEGINF_;
      cmax = fmaxf(cmax, ev[s]);
    }
    cmax = fmaxf(cmax, __shfl_xor(cmax, 16, 32));   // combine lane halves (same row)
    const float nmax = fmaxf(runmax, cmax);
    const float scale = __expf(runmax - nmax);       // first iter: exp(-inf)=0
    float csum = 0.f;
    AFrag pa;
#pragma unroll
    for (int s = 0; s < 16; ++s) {
      const float p = __expf(ev[s] - nmax);
      csum += p;
      pa.h[s] = (_Float16)p;                         // pack straight into A layout
    }
    csum += __shfl_xor(csum, 16, 32);
    runsum = runsum * scale + csum;
    runmax = nmax;

    v8f sv;                                          // per-row rescale broadcast
#pragma unroll
    for (int r = 0; r < 8; ++r) sv[r] = __shfl(scale, r + 8 * half, 32);
#pragma unroll
    for (int nt = 0; nt < 4; ++nt) acc[nt] *= sv;

    const _Float16* buf = mytile + cur * 64 * TP;
#pragma unroll
    for (int nt = 0; nt < 4; ++nt) {                 // B = Wh[j0:j0+32, nt*16..]
      AFrag bf;
      const _Float16* src = buf + (nt * 16 + mrow) * TP + kbase;
      bf.p[0] = *(const v8h*)src;
      bf.p[1] = *(const v8h*)(src + 16);
      acc[nt] = wmma_f16(pa.v, bf.v, acc[nt]);
    }
  }

  v8f inv;
#pragma unroll
  for (int r = 0; r < 8; ++r) inv[r] = 1.f / __shfl(runsum, r + 8 * half, 32);
#pragma unroll
  for (int nt = 0; nt < 4; ++nt)
#pragma unroll
    for (int r = 0; r < 8; ++r) {
      float v = acc[nt][r] * inv[r];
      v = v > 0.f ? v : __expf(v) - 1.f;             // ELU
      const int node = row0 + r + 8 * half;
      hcat[((size_t)b * N_ + node) * HF + h * FH + nt * 16 + mrow] = (_Float16)v;
    }
}

// ---------------------------------------------------------------------------
// Kernel 4: WoutT f16 prep.  Kernel 5: Wh2 = h @ Wout (WMMA), store Wh2T f16.
// ---------------------------------------------------------------------------
__global__ __launch_bounds__(256) void k_prep(const float* __restrict__ wout,
                                              _Float16* __restrict__ woutT) {
  const int i = blockIdx.x * 256 + threadIdx.x;      // HF*C = 16384
  const int f = i >> 5, c = i & 31;
  woutT[(size_t)c * HF + f] = (_Float16)wout[(size_t)f * C_ + c];
}

__global__ __launch_bounds__(256) void k_wh2(const _Float16* __restrict__ hcat,
                                             const _Float16* __restrict__ woutT,
                                             _Float16* __restrict__ wh2T) {
  const int b = blockIdx.y;
  const int n0 = blockIdx.x * 64;
  const int wave = threadIdx.x >> 5, lane = threadIdx.x & 31;
  const int half = lane >> 4, mrow = lane & 15, kbase = half * 8;
  const int mt = wave >> 1, nt = wave & 1;           // 4x2 tiles, one per wave
  const int row = n0 + mt * 16 + mrow;
  const _Float16* __restrict__ arow = hcat + (size_t)(b * N_ + row) * HF;
  const _Float16* __restrict__ brow = woutT + (size_t)(nt * 16 + mrow) * HF;
  v8f acc = {};
#pragma unroll
  for (int ks = 0; ks < 16; ++ks) {                  // K = 512
    const int k0 = ks * 32 + kbase;
    AFrag a, bf;
    a.p[0]  = *(const v8h*)(arow + k0);
    a.p[1]  = *(const v8h*)(arow + k0 + 16);
    bf.p[0] = *(const v8h*)(brow + k0);
    bf.p[1] = *(const v8h*)(brow + k0 + 16);
    acc = wmma_f16(a.v, bf.v, acc);
  }
#pragma unroll
  for (int r = 0; r < 8; ++r) {
    const int node = n0 + mt * 16 + r + 8 * half;
    wh2T[((size_t)b * C_ + nt * 16 + mrow) * N_ + node] = (_Float16)acc[r];
  }
}

// ---------------------------------------------------------------------------
// Kernel 6: g1/g2 = Wh2 . ao1/ao2
// ---------------------------------------------------------------------------
__global__ __launch_bounds__(256) void k_g12(const _Float16* __restrict__ wh2T,
                                             const float* __restrict__ ao1,
                                             const float* __restrict__ ao2,
                                             float* __restrict__ g1,
                                             float* __restrict__ g2) {
  const int b = blockIdx.y;
  const int n = blockIdx.x * 256 + threadIdx.x;
  float s1 = 0.f, s2 = 0.f;
#pragma unroll 8
  for (int c = 0; c < C_; ++c) {
    const float v = (float)wh2T[((size_t)b * C_ + c) * N_ + n];
    s1 += v * ao1[c];
    s2 += v * ao2[c];
  }
  g1[(size_t)b * N_ + n] = s1;
  g2[(size_t)b * N_ + n] = s2;
}

// ---------------------------------------------------------------------------
// Kernel 7: output attention layer (fused masked softmax + P@Wh2, leaky out).
// Same async double-buffered LDS staging, 32x32 f16 tiles (4 transfers each).
// ---------------------------------------------------------------------------
__global__ __launch_bounds__(128) void k_attn2(const _Float16* __restrict__ wh2T,
                                               const float* __restrict__ adj,
                                               const float* __restrict__ g1,
                                               const float* __restrict__ g2,
                                               float* __restrict__ out) {
  __shared__ __align__(16) _Float16 tiles2[4 * 2 * 32 * TP];   // 24 KB
  const int b = blockIdx.y;
  const int wave = threadIdx.x >> 5, lane = threadIdx.x & 31;
  const int row0 = blockIdx.x * 64 + wave * 16;
  const int half = lane >> 4, mrow = lane & 15, kbase = half * 8;
  const int myrow = row0 + mrow;

  const float g1v = g1[(size_t)b * N_ + myrow];
  const float* __restrict__ adjr = adj + ((size_t)b * N_ + myrow) * N_;
  const float* __restrict__ g2r  = g2 + (size_t)b * N_;
  const _Float16* __restrict__ wT = wh2T + (size_t)b * C_ * N_;
  _Float16* mytile = &tiles2[wave * 2 * 32 * TP];

  const int so = lane >> 2;
  const int sseg = lane & 3;

  auto stage = [&](int j0, int bufsel) {   // async copy Wh2T[:, j0:j0+32]
    _Float16* dst = mytile + bufsel * 32 * TP;
#pragma unroll
    for (int c = 0; c < 4; ++c) {
      const int o = c * 8 + so;
      async_b128(lds_addr(dst + o * TP + sseg * 8),
                 wT + (size_t)o * N_ + j0 + sseg * 8);
    }
  };

  v8f acc[2] = {{}, {}};
  float runmax = -INFINITY, runsum = 0.f;

  stage(0, 0);
  for (int j0 = 0; j0 < N_; j0 += 32) {
    const int cur = (j0 >> 5) & 1;
    if (j0 + 32 < N_) {
      wait_ds_0();
      stage(j0 + 32, cur ^ 1);
      asm volatile("s_wait_asynccnt 0x4" ::: "memory");  // 4 newest = next tile
    } else {
      wait_async_0();
    }

    const int cb = j0 + kbase;
    __builtin_prefetch((const void*)(adjr + j0 + 64), 0, 0);
    float av[16], fv[16];
    *(v4f*)&av[0]  = *(const v4f*)(adjr + cb);
    *(v4f*)&av[4]  = *(const v4f*)(adjr + cb + 4);
    *(v4f*)&av[8]  = *(const v4f*)(adjr + cb + 16);
    *(v4f*)&av[12] = *(const v4f*)(adjr + cb + 20);
    *(v4f*)&fv[0]  = *(const v4f*)(g2r + cb);
    *(v4f*)&fv[4]  = *(const v4f*)(g2r + cb + 4);
    *(v4f*)&fv[8]  = *(const v4f*)(g2r + cb + 16);
    *(v4f*)&fv[12] = *(const v4f*)(g2r + cb + 20);

    float ev[16], cmax = -INFINITY;
#pragma unroll
    for (int s = 0; s < 16; ++s) {
      float t = g1v + fv[s];
      t = t > 0.f ? t : ALPHA_ * t;
      ev[s] = av[s] > 0.f ? t : NEGINF_;
      cmax = fmaxf(cmax, ev[s]);
    }
    cmax = fmaxf(cmax, __shfl_xor(cmax, 16, 32));
    const float nmax = fmaxf(runmax, cmax);
    const float scale = __expf(runmax - nmax);
    float csum = 0.f;
    AFrag pa;
#pragma unroll
    for (int s = 0; s < 16; ++s) {
      const float p = __expf(ev[s] - nmax);
      csum += p;
      pa.h[s] = (_Float16)p;
    }
    csum += __shfl_xor(csum, 16, 32);
    runsum = runsum * scale + csum;
    runmax = nmax;

    v8f sv;
#pragma unroll
    for (int r = 0; r < 8; ++r) sv[r] = __shfl(scale, r + 8 * half, 32);
    acc[0] *= sv;
    acc[1] *= sv;

    const _Float16* buf = mytile + cur * 32 * TP;
#pragma unroll
    for (int nt = 0; nt < 2; ++nt) {
      AFrag bf;
      const _Float16* src = buf + (nt * 16 + mrow) * TP + kbase;
      bf.p[0] = *(const v8h*)src;
      bf.p[1] = *(const v8h*)(src + 16);
      acc[nt] = wmma_f16(pa.v, bf.v, acc[nt]);
    }
  }

  v8f inv;
#pragma unroll
  for (int r = 0; r < 8; ++r) inv[r] = 1.f / __shfl(runsum, r + 8 * half, 32);
#pragma unroll
  for (int nt = 0; nt < 2; ++nt)
#pragma unroll
    for (int r = 0; r < 8; ++r) {
      float v = acc[nt][r] * inv[r];
      v = v > 0.f ? v : OUTSLOPE_ * v;               // final leaky_relu
      const int node = row0 + r + 8 * half;
      out[((size_t)b * N_ + node) * C_ + nt * 16 + mrow] = v;
    }
}

// ---------------------------------------------------------------------------
extern "C" void kernel_launch(void* const* d_in, const int* in_sizes, int n_in,
                              void* d_out, int out_size, void* d_ws, size_t ws_size,
                              hipStream_t stream) {
  (void)in_sizes; (void)n_in; (void)out_size; (void)ws_size;
  const float* x    = (const float*)d_in[0];
  const float* adj  = (const float*)d_in[1];
  const float* W    = (const float*)d_in[2];
  const float* a1   = (const float*)d_in[3];
  const float* a2   = (const float*)d_in[4];
  const float* Wout = (const float*)d_in[5];
  const float* ao1  = (const float*)d_in[6];
  const float* ao2  = (const float*)d_in[7];
  float* out = (float*)d_out;

  char* ws = (char*)d_ws;
  size_t off = 0;
  auto alloc = [&](size_t bytes) -> void* {
    void* p = ws + off;
    off = (off + bytes + 255) & ~(size_t)255;
    return p;
  };
  _Float16* whT   = (_Float16*)alloc((size_t)B_ * H_ * FH * N_ * 2);  // 8 MB
  float*    f1    = (float*)alloc((size_t)B_ * H_ * N_ * 4);
  float*    f2    = (float*)alloc((size_t)B_ * H_ * N_ * 4);
  _Float16* hcat  = (_Float16*)alloc((size_t)B_ * N_ * HF * 2);       // 8 MB
  _Float16* woutT = (_Float16*)alloc((size_t)HF * C_ * 2);
  _Float16* wh2T  = (_Float16*)alloc((size_t)B_ * C_ * N_ * 2);
  float*    g1    = (float*)alloc((size_t)B_ * N_ * 4);
  float*    g2    = (float*)alloc((size_t)B_ * N_ * 4);

  k_wh   <<<dim3(N_ / 64, B_ * H_), 256, 0, stream>>>(x, W, whT);
  k_f12  <<<dim3(N_ / 256, B_ * H_), 256, 0, stream>>>(whT, a1, a2, f1, f2);
  k_attn1<<<dim3(N_ / 16, B_), 256, 0, stream>>>(whT, adj, f1, f2, hcat);
  k_prep <<<dim3((HF * C_) / 256), 256, 0, stream>>>(Wout, woutT);
  k_wh2  <<<dim3(N_ / 64, B_), 256, 0, stream>>>(hcat, woutT, wh2T);
  k_g12  <<<dim3(N_ / 256, B_), 256, 0, stream>>>(wh2T, ao1, ao2, g1, g2);
  k_attn2<<<dim3(N_ / 64, B_), 128, 0, stream>>>(wh2T, adj, g1, g2, out);
}
